// MyModel_44822278701241
// MI455X (gfx1250) — matrix operand order repaired and checked by reading, not compiled, not run on hardware
//
#include <hip/hip_runtime.h>
#include <hip/hip_bf16.h>
#include <math.h>

typedef __attribute__((ext_vector_type(16))) _Float16 v16h;
typedef __attribute__((ext_vector_type(8)))  _Float16 v8h;
typedef __attribute__((ext_vector_type(8)))  float    v8f;

typedef __attribute__((ext_vector_type(4))) unsigned int u32x4;
typedef __attribute__((ext_vector_type(8))) int          i32x8;
typedef __attribute__((ext_vector_type(4))) int          i32x4;

#define BATCH 16
#define CH    32   // channels / filters for shared convs

// -------------------- utility kernels --------------------

__global__ void fill_zero_h(_Float16* p, int n) {
  int i = blockIdx.x * blockDim.x + threadIdx.x;
  if (i < n) p[i] = (_Float16)0.f;
}

__global__ void cvt_f32_to_h(const float* __restrict__ s, _Float16* __restrict__ d, int n) {
  int i = blockIdx.x * blockDim.x + threadIdx.x;
  if (i < n) d[i] = (_Float16)s[i];
}

// Wd [8192][10] f32  ->  WdT [16][8192] f16 (rows 10..15 zero)
__global__ void wd_transpose(const float* __restrict__ Wd, _Float16* __restrict__ Wt) {
  int i = blockIdx.x * blockDim.x + threadIdx.x;
  if (i >= 16 * 8192) return;
  int k = i & 8191;
  int n = i >> 13;
  Wt[(size_t)n * 8192 + k] = (_Float16)((n < 10) ? Wd[(size_t)k * 10 + n] : 0.f);
}

// -------------------- conv0: 3x3, Cin=3 -> 32, SAME, ReLU (scalar, tiny) ----
__global__ void conv0_kernel(const float* __restrict__ x, const float* __restrict__ W0,
                             const float* __restrict__ b0, _Float16* __restrict__ Xp) {
  int idx = blockIdx.x * blockDim.x + threadIdx.x;
  if (idx >= BATCH * 64 * 64 * CH) return;
  int f = idx & 31;
  int t = idx >> 5;
  int wc = t & 63; t >>= 6;
  int h  = t & 63;
  int b  = t >> 6;
  float acc = b0[f];
  #pragma unroll
  for (int ki = 0; ki < 3; ++ki) {
    int hi = h + ki - 1;
    if (hi < 0 || hi >= 64) continue;
    #pragma unroll
    for (int kj = 0; kj < 3; ++kj) {
      int wi = wc + kj - 1;
      if (wi < 0 || wi >= 64) continue;
      const float* px = x + (((size_t)b * 64 + hi) * 64 + wi) * 3;
      const float* pw = W0 + ((ki * 3 + kj) * 3) * CH + f;
      acc += px[0] * pw[0] + px[1] * pw[CH] + px[2] * pw[2 * CH];
    }
  }
  acc = acc > 0.f ? acc : 0.f;
  Xp[(((size_t)b * 66 + (h + 1)) * 66 + (wc + 1)) * CH + f] = (_Float16)acc;
}

// -------------------- weight staging into LDS -------------------------------
// Preferred: Tensor Data Mover DMA (tensor_load_to_lds + s_wait_tensorcnt).
// D# built per CDNA5 ISA 8.3/8.4: 1-D tile of 2304 x 8B (= 32x288 f16).
// Fallback: cooperative 16B vector copy.
__device__ __forceinline__ void stage_weights_lds(_Float16* wlds, const _Float16* Wf) {
#if __has_builtin(__builtin_amdgcn_tensor_load_to_lds)
  if ((threadIdx.x >> 5) == 0) {   // wave 0 issues the DMA (TENSOR ops issue per-wave)
    const unsigned long long ga = (unsigned long long)(size_t)Wf;
    const unsigned lds_off = (unsigned)(size_t)(void*)wlds;   // low 32b = LDS offset
    constexpr unsigned N8 = (CH * 288 * 2) / 8;               // 2304 8-byte units

    u32x4 g0;
    g0[0] = 1u;                                   // count=1 (valid descriptor)
    g0[1] = lds_off;                              // lds_addr  [63:32]
    g0[2] = (unsigned)ga;                         // global_addr [95:64]
    g0[3] = (unsigned)((ga >> 32) & 0x1FFFFFFu)   // global_addr [120:96]
            | (2u << 30);                         // type=2 ("image") [127:126]

    i32x8 g1 = {};
    g1[0] = (int)(3u << 16);                      // workgroup_mask=0, data_size=3 (8B)
    g1[1] = (int)((N8 & 0xFFFFu) << 16);          // tensor_dim0[15:0] @ bits[63:48]
    g1[2] = (int)(((N8 >> 16) & 0xFFFFu)          // tensor_dim0[31:16] @ bits[79:64]
                  | (1u << 16));                  // tensor_dim1=1 @ bits[111:80]
    g1[3] = (int)((N8 & 0xFFFFu) << 16);          // tile_dim0 @ bits[127:112]
    g1[4] = 0;                                    // tile_dim1=0 (unused), tile_dim2=0
    g1[5] = (int)N8;                              // tensor_dim0_stride[31:0] @ [191:160]
    g1[6] = 0;                                    // stride hi, tensor_dim1_stride lo
    g1[7] = 0;

    i32x4 g2 = {};                                // dims 2/3 unused (zeros)
    i32x4 g3 = {};
#if defined(__clang_major__) && (__clang_major__ >= 23)
    i32x8 gx = {};
    __builtin_amdgcn_tensor_load_to_lds(g0, g1, g2, g3, gx, 0);
#else
    __builtin_amdgcn_tensor_load_to_lds(g0, g1, g2, g3, 0);
#endif
    __builtin_amdgcn_s_wait_tensorcnt(0);
  }
#else
  {
    const uint4* src = (const uint4*)Wf;
    uint4* dst = (uint4*)wlds;
    constexpr int NV = (CH * 288 * 2) / 16;       // 1152 x 16B
    for (int i = threadIdx.x; i < NV; i += 128) dst[i] = src[i];
  }
#endif
  __syncthreads();
}

// -------------------- shared 3x3 conv via WMMA ------------------------------
// Compile-time H/W so every tap offset folds into load immediates.
// Weights staged once per block into LDS (ds_load_b128 B-fragment reads).
// Each wave computes TWO adjacent 16-position tiles (32 spatial x 32 filters):
// per tap: 2 A fragments + 2 B fragments -> 4 x v_wmma_f32_16x16x32_f16.
template<int H, int W>
__global__ __launch_bounds__(128) void wconv3x3_kernel(
    const _Float16* __restrict__ Xp, const _Float16* __restrict__ Wf,
    const float* __restrict__ bias, _Float16* __restrict__ Yp) {
  constexpr int Hp = H + 2, Wp = W + 2;
  constexpr int tilesPerRow = W / 32;               // double-width tiles

  __shared__ _Float16 wlds[CH * 288];               // 18 KB of 320 KB WGP LDS
  stage_weights_lds(wlds, Wf);

  const int wave = blockIdx.x * 4 + (threadIdx.x >> 5);  // grid sized exactly
  const int lane = threadIdx.x & 31;
  const int tw = wave % tilesPerRow;
  const int th = (wave / tilesPerRow) % H;
  const int b  = wave / (tilesPerRow * H);

  const int m    = lane & 15;                       // A row M / B col N / epilogue N
  const int half = lane >> 4;                       // K-half selector

  // Single base pointer; all 9 tap offsets are compile-time immediates.
  const _Float16* abase =
      Xp + (((size_t)b * Hp + th) * Wp + (tw * 32 + m)) * CH + half * 8;

  v8f acc00 = {}, acc01 = {}, acc10 = {}, acc11 = {};

  #pragma unroll
  for (int t = 0; t < 9; ++t) {
    const int dh = t / 3, dw = t % 3;
    const _Float16* ar0 = abase + ((size_t)dh * Wp + dw) * CH;  // tile 0
    const _Float16* ar1 = ar0 + 16 * CH;                        // tile 1
    v16h a0, a1;
    ((v8h*)&a0)[0] = *(const v8h*)(ar0);        // K 0..7   / 8..15
    ((v8h*)&a0)[1] = *(const v8h*)(ar0 + 16);   // K 16..23 / 24..31
    ((v8h*)&a1)[0] = *(const v8h*)(ar1);
    ((v8h*)&a1)[1] = *(const v8h*)(ar1 + 16);

    const _Float16* b0p = wlds + (size_t)m * 288 + t * 32 + half * 8;
    const _Float16* b1p = b0p + 16 * 288;
    v16h bf0, bf1;
    ((v8h*)&bf0)[0] = *(const v8h*)(b0p);       // K 0..7 / 8..15, then 16..23 / 24..31
    ((v8h*)&bf0)[1] = *(const v8h*)(b0p + 16);
    ((v8h*)&bf1)[0] = *(const v8h*)(b1p);
    ((v8h*)&bf1)[1] = *(const v8h*)(b1p + 16);

    acc00 = __builtin_amdgcn_wmma_f32_16x16x32_f16(false, a0, false, bf0,
                                                   (short)0, acc00, false, false);
    acc01 = __builtin_amdgcn_wmma_f32_16x16x32_f16(false, a0, false, bf1,
                                                   (short)0, acc01, false, false);
    acc10 = __builtin_amdgcn_wmma_f32_16x16x32_f16(false, a1, false, bf0,
                                                   (short)0, acc10, false, false);
    acc11 = __builtin_amdgcn_wmma_f32_16x16x32_f16(false, a1, false, bf1,
                                                   (short)0, acc11, false, false);
  }

  // Epilogue: bias + ReLU + f16 stores into padded interior.
  // C/D layout: lane holds N=(lane&15); VGPR r holds row M = (lane>=16)*8 + r.
  const float bv0 = bias[m];
  const float bv1 = bias[m + 16];
  #pragma unroll
  for (int r = 0; r < 8; ++r) {
    const int M = half * 8 + r;
    _Float16* o0 =
        Yp + (((size_t)b * Hp + (th + 1)) * Wp + (tw * 32 + M + 1)) * CH;
    _Float16* o1 = o0 + 16 * CH;
    float v;
    v = acc00[r] + bv0; o0[m]      = (_Float16)(v > 0.f ? v : 0.f);
    v = acc01[r] + bv1; o0[m + 16] = (_Float16)(v > 0.f ? v : 0.f);
    v = acc10[r] + bv0; o1[m]      = (_Float16)(v > 0.f ? v : 0.f);
    v = acc11[r] + bv1; o1[m + 16] = (_Float16)(v > 0.f ? v : 0.f);
  }
}

// -------------------- 2x2 maxpool stride 2 ---------------------------------
__global__ void maxpool_kernel(const _Float16* __restrict__ in, _Float16* __restrict__ out,
                               int B, int H, int opad) {
  const int Ho = H >> 1, Wo = H >> 1;
  const int Wpi = H + 2;
  const int n = B * Ho * Wo * CH;
  int idx = blockIdx.x * blockDim.x + threadIdx.x;
  if (idx >= n) return;
  int c = idx & 31;
  int t = idx >> 5;
  int ow = t % Wo; t /= Wo;
  int oh = t % Ho;
  int b  = t / Ho;
  const _Float16* base =
      in + (((size_t)b * (H + 2) + (2 * oh + 1)) * Wpi + (2 * ow + 1)) * CH + c;
  float m0 = (float)base[0];
  float m1 = (float)base[CH];
  float m2 = (float)base[(size_t)Wpi * CH];
  float m3 = (float)base[(size_t)Wpi * CH + CH];
  float mv = fmaxf(fmaxf(m0, m1), fmaxf(m2, m3));
  size_t oaddr;
  if (opad) {
    int Wpo = Ho + 2;
    oaddr = (((size_t)b * Wpo + (oh + 1)) * Wpo + (ow + 1)) * CH + c;
  } else {
    oaddr = (((size_t)b * Ho + oh) * Wo + ow) * CH + c;
  }
  out[oaddr] = (_Float16)mv;
}

// -------------------- Dense 16x8192 @ 8192x16 via WMMA ----------------------
// Fragments for a 4-chunk group are loaded into distinct registers first so
// loadcnt can drain progressively instead of a full stall per WMMA.
__global__ void dense_wmma_kernel(const _Float16* __restrict__ X,
                                  const _Float16* __restrict__ Wt,
                                  const float* __restrict__ bd,
                                  float* __restrict__ logits) {
  const int lane = threadIdx.x & 31;
  const int m    = lane & 15;
  const int half = lane >> 4;
  const _Float16* xrow = X  + (size_t)m * 8192;
  const _Float16* wrow = Wt + (size_t)m * 8192;
  v8f acc[4] = {{}, {}, {}, {}};
  for (int kc = 0; kc < 256; kc += 4) {
    v16h a[4], bm[4];
    #pragma unroll
    for (int u = 0; u < 4; ++u) {
      const _Float16* xa = xrow + (size_t)(kc + u) * 32;
      const _Float16* wa = wrow + (size_t)(kc + u) * 32;
      ((v8h*)&a[u])[0]  = *(const v8h*)(xa + half * 8);
      ((v8h*)&a[u])[1]  = *(const v8h*)(xa + 16 + half * 8);
      ((v8h*)&bm[u])[0] = *(const v8h*)(wa + half * 8);
      ((v8h*)&bm[u])[1] = *(const v8h*)(wa + 16 + half * 8);
    }
    #pragma unroll
    for (int u = 0; u < 4; ++u)
      acc[u] = __builtin_amdgcn_wmma_f32_16x16x32_f16(false, a[u], false, bm[u],
                                                      (short)0, acc[u], false, false);
  }
  const float bv = (m < 10) ? bd[m] : 0.f;
  #pragma unroll
  for (int r = 0; r < 8; ++r) {
    const int M = half * 8 + r;  // batch row
    logits[M * 16 + m] = acc[0][r] + acc[1][r] + acc[2][r] + acc[3][r] + bv;
  }
}

// -------------------- softmax over 10 logits per batch ----------------------
__global__ void softmax_kernel(const float* __restrict__ logits, float* __restrict__ out) {
  int b = blockIdx.x * blockDim.x + threadIdx.x;
  if (b >= BATCH) return;
  float mx = -1e30f;
  for (int j = 0; j < 10; ++j) mx = fmaxf(mx, logits[b * 16 + j]);
  float e[10], s = 0.f;
  for (int j = 0; j < 10; ++j) { e[j] = expf(logits[b * 16 + j] - mx); s += e[j]; }
  float inv = 1.f / s;
  for (int j = 0; j < 10; ++j) out[b * 10 + j] = e[j] * inv;
}

// -------------------- host-side orchestration -------------------------------

static inline size_t align256(size_t x) { return (x + 255) & ~(size_t)255; }

extern "C" void kernel_launch(void* const* d_in, const int* in_sizes, int n_in,
                              void* d_out, int out_size, void* d_ws, size_t ws_size,
                              hipStream_t stream) {
  (void)in_sizes; (void)n_in; (void)out_size; (void)ws_size;

  const float* x   = (const float*)d_in[0];  // [16][64][64][3]
  const float* W0  = (const float*)d_in[1];  // [3][3][3][32] HWIO
  const float* b0  = (const float*)d_in[2];  // [32]
  const float* w   = (const float*)d_in[3];  // [32][288]  (f, tap*32+c)
  const float* bsh = (const float*)d_in[4];  // [32]
  const float* Wd  = (const float*)d_in[5];  // [8192][10]
  const float* bd  = (const float*)d_in[6];  // [10]
  float* out = (float*)d_out;                // [16][10]

  const size_t nXp = (size_t)BATCH * 66 * 66 * CH;   // padded 64x64 stage
  const size_t nSp = (size_t)BATCH * 34 * 34 * CH;   // padded 32x32 stage
  const size_t nP2 = (size_t)BATCH * 16 * 16 * CH;   // tight pooled output
  const size_t nWf = (size_t)CH * 288;
  const size_t nWt = (size_t)16 * 8192;

  char* p = (char*)d_ws;
  _Float16* Xp = (_Float16*)p; p += align256(nXp * 2);
  _Float16* Yp = (_Float16*)p; p += align256(nXp * 2);
  _Float16* Sp = (_Float16*)p; p += align256(nSp * 2);
  _Float16* Tp = (_Float16*)p; p += align256(nSp * 2);
  _Float16* P2 = (_Float16*)p; p += align256(nP2 * 2);
  _Float16* Wf = (_Float16*)p; p += align256(nWf * 2);
  _Float16* Wt = (_Float16*)p; p += align256(nWt * 2);
  float*    lg = (float*)p;    p += align256(16 * 16 * sizeof(float));

  const int TB = 256;
  #define GRID(n) ((int)(((n) + TB - 1) / TB))

  // 1) zero padded buffers (halos must be 0 every call; deterministic re-run)
  fill_zero_h<<<GRID(nXp), TB, 0, stream>>>(Xp, (int)nXp);
  fill_zero_h<<<GRID(nXp), TB, 0, stream>>>(Yp, (int)nXp);
  fill_zero_h<<<GRID(nSp), TB, 0, stream>>>(Sp, (int)nSp);
  fill_zero_h<<<GRID(nSp), TB, 0, stream>>>(Tp, (int)nSp);

  // 2) weight conversions
  cvt_f32_to_h<<<GRID(nWf), TB, 0, stream>>>(w, Wf, (int)nWf);
  wd_transpose<<<GRID(nWt), TB, 0, stream>>>(Wd, Wt);

  // 3) conv0 + relu -> Xp (padded f16)
  conv0_kernel<<<GRID((size_t)BATCH * 64 * 64 * CH), TB, 0, stream>>>(x, W0, b0, Xp);

  // 4) shared convs via WMMA (4 waves / 128-thread block; exact grids)
  {
    const int nW64 = BATCH * 64 * (64 / 32);   // 2048 waves -> 512 blocks
    const int nW32 = BATCH * 32 * (32 / 32);   // 512 waves  -> 128 blocks
    wconv3x3_kernel<64, 64><<<nW64 / 4, 128, 0, stream>>>(Xp, Wf, bsh, Yp); // conv1
    wconv3x3_kernel<64, 64><<<nW64 / 4, 128, 0, stream>>>(Yp, Wf, bsh, Xp); // conv2
    maxpool_kernel<<<GRID((size_t)BATCH * 32 * 32 * CH), TB, 0, stream>>>(Xp, Sp, BATCH, 64, 1);
    wconv3x3_kernel<32, 32><<<nW32 / 4, 128, 0, stream>>>(Sp, Wf, bsh, Tp); // conv3
    wconv3x3_kernel<32, 32><<<nW32 / 4, 128, 0, stream>>>(Tp, Wf, bsh, Sp); // conv4
    maxpool_kernel<<<GRID((size_t)BATCH * 16 * 16 * CH), TB, 0, stream>>>(Sp, P2, BATCH, 32, 0);
  }

  // 5) dense (single wave) + softmax
  dense_wmma_kernel<<<1, 32, 0, stream>>>(P2, Wt, bd, lg);
  softmax_kernel<<<1, 32, 0, stream>>>(lg, out);
  #undef GRID
}
